// FrameTransformerEncoder_18648747999563
// MI455X (gfx1250) — matrix operand order repaired and checked by reading, not compiled, not run on hardware
//
#include <hip/hip_runtime.h>
#include <hip/hip_bf16.h>

// ---------------------------------------------------------------------------
// FrameTransformerEncoder forward for MI455X (gfx1250, wave32, WMMA).
// All matmuls run through v_wmma_f32_16x16x32_bf16 (bf16 in, f32 accum):
// bf16 WMMA has 8x the K-depth/instr of fp32 WMMA, putting this
// ~38 GFLOP / ~250 MB workload at the 23.3 TB/s memory roofline.
// GEMM: 128 threads (4 waves), block tile 128x64, wave tile 64x32
// (4x2 accumulators -> 8 WMMA per 12 ds_load_b128 per K-step of 32),
// software-pipelined with LDS double buffering: next tile's global loads
// are in flight while the current tile's WMMAs execute.
// ---------------------------------------------------------------------------

typedef __attribute__((ext_vector_type(16))) __bf16 v16bf;
typedef __attribute__((ext_vector_type(8)))  __bf16 v8bf;
typedef __attribute__((ext_vector_type(8)))  float  v8f;

#define F_  512
#define W_  256
#define C_  8
#define CO_ 8
#define NH_ 8
#define DH_ 64
#define CC_ 16
#define FC_ 24
#define FH_ 2048

__device__ __forceinline__ unsigned short f2bf_rne(float f) {
  unsigned u = __float_as_uint(f);
  u += 0x7FFFu + ((u >> 16) & 1u);
  return (unsigned short)(u >> 16);
}

union FragU { v16bf v16; v8bf v8[2]; };

__device__ __forceinline__ v16bf mk_frag(const unsigned short* p0,
                                         const unsigned short* p1) {
  FragU f;
  f.v8[0] = *(const v8bf*)(const void*)p0;   // 16B-aligned LDS reads
  f.v8[1] = *(const v8bf*)(const void*)p1;
  return f.v16;
}

// ---------------------------------------------------------------------------
// Batched GEMM: C[z](M,N) = alpha * A[z](M,K) * B[z](K,N)
// element(A,m,k) = A[z*aB + m*aR + k*aC] (f32 in memory, bf16 in LDS/WMMA).
// ACONT: aC==1 (rows contiguous)  BCONT: bC==1 (rows contiguous).
// All strides are 32-bit: every buffer here is < 2^31 elements.
// ---------------------------------------------------------------------------
#define TM 128
#define TN 64
#define TK 32
#define LDP 40   // LDS row pitch (ushorts): 80B rows keep 16B chunk alignment

template <bool ACONT, bool BCONT>
__global__ __launch_bounds__(128)
void gemm_k(const float* __restrict__ A, const float* __restrict__ B,
            float* __restrict__ Cp,
            int K,
            int aB, int aR, int aC,
            int bB, int bR, int bC,
            int cB, int cR, int cC,
            float alpha)
{
  __shared__ __attribute__((aligned(16))) unsigned short At[2][TM * LDP];
  __shared__ __attribute__((aligned(16))) unsigned short Bt[2][TN * LDP];

  const int tid  = threadIdx.x;
  const int lane = tid & 31;
  const int wv   = tid >> 5;
  const int wm   = wv >> 1;          // wave row (2 x 64 rows)
  const int wn   = wv & 1;           // wave col (2 x 32 cols)
  const int hh   = lane >> 4;        // half-wave select (K split)
  const int ll   = lane & 15;        // row / col within 16

  const int aBase = blockIdx.z * aB;
  const int bBase = blockIdx.z * bB;
  const int m0    = blockIdx.y * TM;
  const int n0    = blockIdx.x * TN;

  v8f acc[4][2];
  #pragma unroll
  for (int i = 0; i < 4; i++)
    #pragma unroll
    for (int j = 0; j < 2; j++) acc[i][j] = (v8f)0.f;

  // staging registers for the in-flight tile
  float4 aR4[ACONT ? 8 : 1];
  float  aRS[ACONT ? 1 : 32];
  float4 bR4[BCONT ? 4 : 1];
  float  bRS[BCONT ? 1 : 16];

  // ---- issue global loads of tile at k0 into registers ----
  auto gload = [&](int k0) {
    if constexpr (ACONT) {
      #pragma unroll
      for (int t = 0; t < 8; t++) {
        int e  = tid + t * 128;            // float4 index over [128][8]
        int r  = e >> 3;
        int c4 = (e & 7) * 4;
        const float* s = A + (aBase + (m0 + r) * aR + (k0 + c4));
        aR4[t] = *(const float4*)s;
        if (k0 + TK < K) __builtin_prefetch((const void*)(s + TK), 0, 3);
      }
    } else {
      #pragma unroll
      for (int t = 0; t < 32; t++) {
        int e = t * 128 + tid;             // r fastest: coalesces when aR==1
        int k = e >> 7;
        int r = e & 127;
        aRS[t] = A[aBase + (m0 + r) * aR + (k0 + k) * aC];
      }
    }
    if constexpr (BCONT) {
      #pragma unroll
      for (int t = 0; t < 4; t++) {
        int e  = tid + t * 128;            // float4 index over [32][16]
        int k  = e >> 4;
        int c4 = (e & 15) * 4;
        const float* s = B + (bBase + (k0 + k) * bR + (n0 + c4));
        bR4[t] = *(const float4*)s;
        if (k0 + TK < K) __builtin_prefetch((const void*)(s + TK * bR), 0, 3);
      }
    } else {
      #pragma unroll
      for (int t = 0; t < 16; t++) {
        int e = t * 128 + tid;             // k fastest: coalesces when bR==1
        int n = e >> 5;
        int k = e & 31;
        bRS[t] = B[bBase + (k0 + k) * bR + (n0 + n) * bC];
      }
    }
  };

  // ---- convert staged registers to bf16 and store into LDS buffer ----
  auto cstore = [&](int buf) {
    unsigned short* Atb = At[buf];
    unsigned short* Btb = Bt[buf];
    if constexpr (ACONT) {
      #pragma unroll
      for (int t = 0; t < 8; t++) {
        int e  = tid + t * 128;
        int r  = e >> 3;
        int c4 = (e & 7) * 4;
        unsigned short* d = &Atb[r * LDP + c4];
        d[0] = f2bf_rne(aR4[t].x); d[1] = f2bf_rne(aR4[t].y);
        d[2] = f2bf_rne(aR4[t].z); d[3] = f2bf_rne(aR4[t].w);
      }
    } else {
      #pragma unroll
      for (int t = 0; t < 32; t++) {
        int e = t * 128 + tid;
        int k = e >> 7;
        int r = e & 127;
        Atb[r * LDP + k] = f2bf_rne(aRS[t]);
      }
    }
    if constexpr (BCONT) {
      #pragma unroll
      for (int t = 0; t < 4; t++) {
        int e  = tid + t * 128;
        int k  = e >> 4;
        int c4 = (e & 15) * 4;
        Btb[(c4 + 0) * LDP + k] = f2bf_rne(bR4[t].x);
        Btb[(c4 + 1) * LDP + k] = f2bf_rne(bR4[t].y);
        Btb[(c4 + 2) * LDP + k] = f2bf_rne(bR4[t].z);
        Btb[(c4 + 3) * LDP + k] = f2bf_rne(bR4[t].w);
      }
    } else {
      #pragma unroll
      for (int t = 0; t < 16; t++) {
        int e = t * 128 + tid;
        int n = e >> 5;
        int k = e & 31;
        Btb[n * LDP + k] = f2bf_rne(bRS[t]);
      }
    }
  };

  // ---- 8 WMMAs from LDS buffer per CDNA5 16-bit fragment layouts ----
  // A 16x32: lane m (both halves) = row m; half0: K 0-7 & 16-23,
  //          half1: K 8-15 & 24-31.
// B 32x16: lane n = column n; half0: K 0-7 & 16-23, half1: K 8-15 & 24-31.
  auto compute = [&](int buf) {
    const unsigned short* Atb = At[buf];
    const unsigned short* Btb = Bt[buf];
    v16bf aF[4], bF[2];
    #pragma unroll
    for (int i = 0; i < 4; i++) {
      const unsigned short* pa = &Atb[(wm * 64 + i * 16 + ll) * LDP + 8 * hh];
      aF[i] = mk_frag(pa, pa + 16);
    }
    #pragma unroll
    for (int j = 0; j < 2; j++) {
      const unsigned short* pb = &Btb[(wn * 32 + j * 16 + ll) * LDP + 8 * hh];
      bF[j] = mk_frag(pb, pb + 16);
    }
    #pragma unroll
    for (int i = 0; i < 4; i++)
      #pragma unroll
      for (int j = 0; j < 2; j++)
        acc[i][j] = __builtin_amdgcn_wmma_f32_16x16x32_bf16(
            false, aF[i], false, bF[j], (short)0, acc[i][j], false, false);
  };

  // ---- software-pipelined main loop (double-buffered LDS) ----
  gload(0);
  cstore(0);
  __syncthreads();
  int cur = 0;
  for (int k0 = TK; k0 < K; k0 += TK) {
    gload(k0);        // next tile's global loads in flight...
    compute(cur);     // ...while current tile's WMMAs execute
    cstore(cur ^ 1);  // waits loads, converts, fills other buffer
    __syncthreads();  // staging done + all waves finished reading `cur`
    cur ^= 1;
  }
  compute(cur);

  // ---- store: C/D 16x16 f32 layout: VGPR r -> M = 8*hh + r, N = ll ----
  const int cBase = blockIdx.z * cB;
  #pragma unroll
  for (int i = 0; i < 4; i++) {
    #pragma unroll
    for (int j = 0; j < 2; j++) {
      int col  = n0 + wn * 32 + j * 16 + ll;
      int rowb = m0 + wm * 64 + i * 16 + hh * 8;
      #pragma unroll
      for (int r = 0; r < 8; r++)
        Cp[cBase + (rowb + r) * cR + col * cC] = acc[i][j][r] * alpha;
    }
  }
}

// ---------------------------------------------------------------------------
// LayerNorm over the F axis, per (c, w). One block per column of 512.
// ---------------------------------------------------------------------------
__global__ __launch_bounds__(256)
void ln_kernel(const float* __restrict__ x, const float* __restrict__ g,
               const float* __restrict__ b, float* __restrict__ h) {
  __shared__ float red[256];
  int c = blockIdx.x / W_;
  int w = blockIdx.x % W_;
  int t = threadIdx.x;
  const float* xp = x + (long)c * F_ * W_ + w;
  float a0 = xp[(long)t * W_];
  float a1 = xp[(long)(t + 256) * W_];
  red[t] = a0 + a1; __syncthreads();
  for (int s = 128; s > 0; s >>= 1) { if (t < s) red[t] += red[t + s]; __syncthreads(); }
  float mu = red[0] * (1.f / 512.f); __syncthreads();
  float d0 = a0 - mu, d1 = a1 - mu;
  red[t] = d0 * d0 + d1 * d1; __syncthreads();
  for (int s = 128; s > 0; s >>= 1) { if (t < s) red[t] += red[t + s]; __syncthreads(); }
  float rs = rsqrtf(red[0] * (1.f / 512.f) + 1e-5f);
  float* hp = h + (long)c * F_ * W_ + w;
  const float* gp = g + c * F_;
  const float* bp = b + c * F_;
  hp[(long)t * W_]         = d0 * rs * gp[t]       + bp[t];
  hp[(long)(t + 256) * W_] = d1 * rs * gp[t + 256] + bp[t + 256];
}

// ---------------------------------------------------------------------------
// frame_conv: Conv2d kernel (1,3), pad (0,1): 3-tap conv along W mixing
// 8 -> 8 channels. x,y: [8][F][W]; cw: [8][8][3]; cb: [8].
// ---------------------------------------------------------------------------
__global__ __launch_bounds__(256)
void fconv_kernel(const float* __restrict__ x, const float* __restrict__ cw,
                  const float* __restrict__ cb, float* __restrict__ y) {
  __shared__ float wsm[8 * 8 * 3];
  __shared__ float bsm[8];
  int t = threadIdx.x;
  if (t < 192) wsm[t] = cw[t];
  if (t < 8)   bsm[t] = cb[t];
  __syncthreads();
  long s = (long)blockIdx.x * 256 + t;       // f*W + w
  int w = (int)(s & (W_ - 1));
  float in[8][3];
  #pragma unroll
  for (int i = 0; i < 8; i++) {
    long base = (long)i * F_ * W_ + s;
    in[i][0] = (w > 0)       ? x[base - 1] : 0.f;
    in[i][1] = x[base];
    in[i][2] = (w < W_ - 1)  ? x[base + 1] : 0.f;
  }
  #pragma unroll
  for (int o = 0; o < 8; o++) {
    float a = bsm[o];
    #pragma unroll
    for (int i = 0; i < 8; i++) {
      const float* wp = &wsm[(o * 8 + i) * 3];
      a = fmaf(in[i][0], wp[0], a);
      a = fmaf(in[i][1], wp[1], a);
      a = fmaf(in[i][2], wp[2], a);
    }
    y[(long)o * F_ * W_ + s] = a;
  }
}

// ---------------------------------------------------------------------------
// Row softmax over 256 logits. Reads logits from d_out (qk are outputs),
// writes probabilities to workspace.
// ---------------------------------------------------------------------------
__global__ __launch_bounds__(256)
void softmax_kernel(const float* __restrict__ S, float* __restrict__ P) {
  __shared__ float red[256];
  long row = blockIdx.x;
  int  t   = threadIdx.x;
  float v = S[row * 256 + t];
  red[t] = v; __syncthreads();
  for (int s = 128; s > 0; s >>= 1) { if (t < s) red[t] = fmaxf(red[t], red[t + s]); __syncthreads(); }
  float m = red[0]; __syncthreads();
  float e = __expf(v - m);
  red[t] = e; __syncthreads();
  for (int s = 128; s > 0; s >>= 1) { if (t < s) red[t] += red[t + s]; __syncthreads(); }
  P[row * 256 + t] = e / red[0];
}

// ---------------------------------------------------------------------------
// Channel mix: y[o][s] = sum_c dw[o][c] * x[c][s], optional exact GELU,
// optional residual add. Bandwidth-bound -> plain VALU.
// ---------------------------------------------------------------------------
template <int CIN, int COUT>
__global__ __launch_bounds__(256)
void mix_kernel(const float* __restrict__ x, const float* __restrict__ dw,
                const float* __restrict__ resid, float* __restrict__ y,
                long S, int gelu) {
  long s = (long)blockIdx.x * 256 + threadIdx.x;
  if (s >= S) return;
  float xi[CIN];
  #pragma unroll
  for (int c = 0; c < CIN; c++) xi[c] = x[(long)c * S + s];
  #pragma unroll
  for (int o = 0; o < COUT; o++) {
    float a = 0.f;
    #pragma unroll
    for (int c = 0; c < CIN; c++) a = fmaf(dw[o * CIN + c], xi[c], a);
    if (gelu) a = 0.5f * a * (1.f + erff(a * 0.70710678118654752f));
    if (resid) a += resid[(long)o * S + s];
    y[(long)o * S + s] = a;
  }
}

// ---------------------------------------------------------------------------
extern "C" void kernel_launch(void* const* d_in, const int* in_sizes, int n_in,
                              void* d_out, int out_size, void* d_ws, size_t ws_size,
                              hipStream_t stream) {
  const float* x       = (const float*)d_in[0];
  const float* src     = (const float*)d_in[1];
  const float* ln_g    = (const float*)d_in[2];
  const float* ln_b    = (const float*)d_in[3];
  const float* sa_q_pw = (const float*)d_in[4];
  const float* sa_q_cw = (const float*)d_in[5];
  const float* sa_q_cb = (const float*)d_in[6];
  const float* sa_k_pw = (const float*)d_in[7];
  const float* sa_k_cw = (const float*)d_in[8];
  const float* sa_k_cb = (const float*)d_in[9];
  const float* sa_v_pw = (const float*)d_in[10];
  const float* sa_v_cw = (const float*)d_in[11];
  const float* sa_v_cb = (const float*)d_in[12];
  const float* sa_o_pw = (const float*)d_in[13];
  const float* ca_q_pw = (const float*)d_in[14];
  const float* ca_q_dw = (const float*)d_in[15];
  const float* ca_q_cw = (const float*)d_in[16];
  const float* ca_q_cb = (const float*)d_in[17];
  const float* ca_k_pw = (const float*)d_in[18];
  const float* ca_k_cw = (const float*)d_in[19];
  const float* ca_k_cb = (const float*)d_in[20];
  const float* ca_v_pw = (const float*)d_in[21];
  const float* ca_v_cw = (const float*)d_in[22];
  const float* ca_v_cb = (const float*)d_in[23];
  const float* ca_o_pw = (const float*)d_in[24];
  const float* ffn1_pw = (const float*)d_in[25];
  const float* ffn1_dw = (const float*)d_in[26];
  const float* ffn2_pw = (const float*)d_in[27];
  const float* ffn2_dw = (const float*)d_in[28];

  float* out = (float*)d_out;
  float* ws  = (float*)d_ws;

  constexpr long FW  = (long)F_ * W_;       // 131072
  constexpr long CFW = (long)CO_ * FW;      // 1,048,576
  constexpr long FF  = (long)F_ * F_;       // 262144
  constexpr int  iFW = (int)FW, iFF = (int)FF;
  // d_out layout: [x+z | qk1 | qk2]
  constexpr long QK1 = CFW;
  constexpr long QK2 = CFW + 64L * 256 * 256;
  // workspace layout (phase-based reuse; ~120 MB total)
  constexpr long OFF_H   = 0;
  constexpr long OFF_SA  = 1 * CFW;
  constexpr long OFF_CA  = 2 * CFW;
  constexpr long OFF_Q   = 3 * CFW;
  constexpr long OFF_K   = 4 * CFW;
  constexpr long OFF_V   = 5 * CFW;
  constexpr long OFF_LIN = 6 * CFW;         // 2*CFW (16ch CA query lin)
  constexpr long OFF_ATT = 8 * CFW;
  constexpr long OFF_P   = 9 * CFW;         // 4*CFW attn probs
  constexpr long OFF_Z1  = 3 * CFW;         // FFN phase: reuse attn scratch
  constexpr long OFF_Z2  = 15 * CFW;
  constexpr long OFF_Z3  = 27 * CFW;

  const float qk_scale = 0.044194173824159216f;   // 1/sqrt(512)

  // ---- 1. h = layernorm(x) ----
  ln_kernel<<<C_ * W_, 256, 0, stream>>>(x, ln_g, ln_b, ws + OFF_H);

  // ---- 2. self-attention ----
  // q/k/v = frame_conv(pw @ h): M=512,N=256,K=512, batch 8
  gemm_k<true, true><<<dim3(4, 4, 8), 128, 0, stream>>>(
      sa_q_pw, ws + OFF_H, ws + OFF_LIN, F_, iFF, F_, 1, iFW, W_, 1, iFW, W_, 1, 1.f);
  fconv_kernel<<<FW / 256, 256, 0, stream>>>(ws + OFF_LIN, sa_q_cw, sa_q_cb, ws + OFF_Q);
  gemm_k<true, true><<<dim3(4, 4, 8), 128, 0, stream>>>(
      sa_k_pw, ws + OFF_H, ws + OFF_LIN, F_, iFF, F_, 1, iFW, W_, 1, iFW, W_, 1, 1.f);
  fconv_kernel<<<FW / 256, 256, 0, stream>>>(ws + OFF_LIN, sa_k_cw, sa_k_cb, ws + OFF_K);
  gemm_k<true, true><<<dim3(4, 4, 8), 128, 0, stream>>>(
      sa_v_pw, ws + OFF_H, ws + OFF_LIN, F_, iFF, F_, 1, iFW, W_, 1, iFW, W_, 1, 1.f);
  fconv_kernel<<<FW / 256, 256, 0, stream>>>(ws + OFF_LIN, sa_v_cw, sa_v_cb, ws + OFF_V);
  // qk1[z](wq,wk) = scale * Qh Kh^T ; batch z = c*NH + h, head base z*DH*W
  // A strided: A(m,k) = q[z*DH*W + k*W + m]
  gemm_k<false, true><<<dim3(4, 2, 64), 128, 0, stream>>>(
      ws + OFF_Q, ws + OFF_K, out + QK1, DH_,
      DH_ * W_, 1, W_, DH_ * W_, W_, 1, 65536, 256, 1, qk_scale);
  softmax_kernel<<<64 * 256, 256, 0, stream>>>(out + QK1, ws + OFF_P);
  // attn = P @ Vh, output written transposed into [f][w] layout
  gemm_k<true, false><<<dim3(1, 2, 64), 128, 0, stream>>>(
      ws + OFF_P, ws + OFF_V, ws + OFF_ATT, W_,
      65536, 256, 1, DH_ * W_, 1, W_, DH_ * W_, 1, W_, 1.f);
  // sa = o_pw @ attn
  gemm_k<true, true><<<dim3(4, 4, 8), 128, 0, stream>>>(
      sa_o_pw, ws + OFF_ATT, ws + OFF_SA, F_, iFF, F_, 1, iFW, W_, 1, iFW, W_, 1, 1.f);

  // ---- 3. cross-attention ----
  gemm_k<true, true><<<dim3(4, 4, 8), 128, 0, stream>>>(
      ca_q_pw, ws + OFF_H, ws + OFF_LIN, F_, iFF, F_, 1, iFW, W_, 1, iFW, W_, 1, 1.f);
  gemm_k<true, true><<<dim3(4, 4, 8), 128, 0, stream>>>(
      ca_q_pw + 8 * FF, ws + OFF_SA, ws + OFF_LIN + 8 * FW, F_,
      iFF, F_, 1, iFW, W_, 1, iFW, W_, 1, 1.f);
  mix_kernel<16, 8><<<FW / 256, 256, 0, stream>>>(ws + OFF_LIN, ca_q_dw, nullptr, ws + OFF_ATT, FW, 0);
  fconv_kernel<<<FW / 256, 256, 0, stream>>>(ws + OFF_ATT, ca_q_cw, ca_q_cb, ws + OFF_Q);
  gemm_k<true, true><<<dim3(4, 4, 8), 128, 0, stream>>>(
      ca_k_pw, src, ws + OFF_LIN, F_, iFF, F_, 1, iFW, W_, 1, iFW, W_, 1, 1.f);
  fconv_kernel<<<FW / 256, 256, 0, stream>>>(ws + OFF_LIN, ca_k_cw, ca_k_cb, ws + OFF_K);
  gemm_k<true, true><<<dim3(4, 4, 8), 128, 0, stream>>>(
      ca_v_pw, src, ws + OFF_LIN, F_, iFF, F_, 1, iFW, W_, 1, iFW, W_, 1, 1.f);
  fconv_kernel<<<FW / 256, 256, 0, stream>>>(ws + OFF_LIN, ca_v_cw, ca_v_cb, ws + OFF_V);
  gemm_k<false, true><<<dim3(4, 2, 64), 128, 0, stream>>>(
      ws + OFF_Q, ws + OFF_K, out + QK2, DH_,
      DH_ * W_, 1, W_, DH_ * W_, W_, 1, 65536, 256, 1, qk_scale);
  softmax_kernel<<<64 * 256, 256, 0, stream>>>(out + QK2, ws + OFF_P);
  gemm_k<true, false><<<dim3(1, 2, 64), 128, 0, stream>>>(
      ws + OFF_P, ws + OFF_V, ws + OFF_ATT, W_,
      65536, 256, 1, DH_ * W_, 1, W_, DH_ * W_, 1, W_, 1.f);
  gemm_k<true, true><<<dim3(4, 4, 8), 128, 0, stream>>>(
      ca_o_pw, ws + OFF_ATT, ws + OFF_CA, F_, iFF, F_, 1, iFW, W_, 1, iFW, W_, 1, 1.f);

  // ---- 4. FFN over concat(h, sa, ca) (24ch) ----
  constexpr long FHF = (long)FH_ * F_;     // 1,048,576
  constexpr long FHW = (long)FH_ * W_;     // 524,288
  constexpr int  iFHF = (int)FHF, iFHW = (int)FHW;
  // M=2048, N=256, K=512, batch 8 per source
  gemm_k<true, true><<<dim3(4, 16, 8), 128, 0, stream>>>(
      ffn1_pw, ws + OFF_H, ws + OFF_Z1, F_,
      iFHF, F_, 1, iFW, W_, 1, iFHW, W_, 1, 1.f);
  gemm_k<true, true><<<dim3(4, 16, 8), 128, 0, stream>>>(
      ffn1_pw + 8 * FHF, ws + OFF_SA, ws + OFF_Z1 + 8 * FHW, F_,
      iFHF, F_, 1, iFW, W_, 1, iFHW, W_, 1, 1.f);
  gemm_k<true, true><<<dim3(4, 16, 8), 128, 0, stream>>>(
      ffn1_pw + 16 * FHF, ws + OFF_CA, ws + OFF_Z1 + 16 * FHW, F_,
      iFHF, F_, 1, iFW, W_, 1, iFHW, W_, 1, 1.f);
  // dw mix 24->24 + exact GELU
  mix_kernel<24, 24><<<(int)(FHW / 256), 256, 0, stream>>>(
      ws + OFF_Z1, ffn1_dw, nullptr, ws + OFF_Z2, FHW, 1);
  // ffn2 pw: [24][F][FH] @ z2 : M=512, N=256, K=2048, batch 24
  gemm_k<true, true><<<dim3(4, 4, 24), 128, 0, stream>>>(
      ffn2_pw, ws + OFF_Z2, ws + OFF_Z3, FH_,
      iFHF, FH_, 1, iFHW, W_, 1, iFW, W_, 1, 1.f);
  // out = x + ffn2_dw(8x24) @ z3
  mix_kernel<24, 8><<<FW / 256, 256, 0, stream>>>(ws + OFF_Z3, ffn2_dw, x, out, FW, 0);
}
